// GQAttention_85512798863962
// MI455X (gfx1250) — matrix operand order repaired and checked by reading, not compiled, not run on hardware
//
#include <hip/hip_runtime.h>
#include <hip/hip_bf16.h>

typedef _Float16 f16;
typedef _Float16 v16h __attribute__((ext_vector_type(16)));
typedef _Float16 v8h  __attribute__((ext_vector_type(8)));
typedef _Float16 v4h  __attribute__((ext_vector_type(4)));
typedef float    v8f  __attribute__((ext_vector_type(8)));
typedef float    v4f  __attribute__((ext_vector_type(4)));

#define WMMA_F32_F16(A, B, C) \
  __builtin_amdgcn_wmma_f32_16x16x32_f16(false, (A), false, (B), (short)0, (C), false, false)

static __device__ __forceinline__ v16h concat8(v8h lo, v8h hi) {
  return __builtin_shufflevector(lo, hi, 0, 1, 2, 3, 4, 5, 6, 7,
                                         8, 9, 10, 11, 12, 13, 14, 15);
}

// Async f16 16B copy global -> LDS (gfx1250 GLOBAL_LOAD_ASYNC_TO_LDS_B128).
static __device__ __forceinline__ void async_copy16(const f16* g, f16* l) {
  unsigned lds = (unsigned)(uintptr_t)l;
  unsigned long long ga = (unsigned long long)(uintptr_t)g;
  asm volatile("global_load_async_to_lds_b128 %0, %1, off"
               :: "v"(lds), "v"(ga) : "memory");
}
static __device__ __forceinline__ void wait_async0() {
  asm volatile("s_wait_asynccnt 0x0" ::: "memory");
}

static constexpr int Bz = 2, T = 2048, H = 2048, NH = 16, NKV = 4, HD = 128;

// ---------------------------------------------------------------------------
// Kernel 1: fused QKV projection.  C[4096, 3072] = x[4096,2048] @ [Wq|Wk|Wv].
// Double-buffered LDS; wide b128 global loads batched into registers.
// ---------------------------------------------------------------------------
__global__ __launch_bounds__(256) void qkv_gemm(
    const float* __restrict__ x, const float* __restrict__ Wq,
    const float* __restrict__ Wk, const float* __restrict__ Wv,
    f16* __restrict__ qb, f16* __restrict__ kb, f16* __restrict__ vb) {
  __shared__ __align__(16) f16 As[2][128 * 32];   // A[m][k]
  __shared__ __align__(16) f16 Bt[2][128 * 32];   // B transposed: Bt[n][k]
  const int tid = threadIdx.x, lane = tid & 31, w = tid >> 5;
  const int wmy = w & 3, wnx = w >> 2;
  const int hl = lane >> 4, ml = lane & 15;
  const int m0 = blockIdx.y * 128, n0 = blockIdx.x * 128;

  const float* W; int ld, cn0;
  if (n0 < 2048)      { W = Wq; ld = 2048; cn0 = n0; }
  else if (n0 < 2560) { W = Wk; ld = 512;  cn0 = n0 - 2048; }
  else                { W = Wv; ld = 512;  cn0 = n0 - 2560; }

  const int ar = tid >> 3, acq = (tid & 7) * 4;     // A quad coords
  const int bk = tid >> 5, bnq = (tid & 31) * 4;    // B quad coords

  v4f areg[4], breg[4];
  auto load_tiles = [&](int kk) {
    #pragma unroll
    for (int i = 0; i < 4; ++i)
      areg[i] = *(const v4f*)&x[(size_t)(m0 + ar + 32 * i) * H + kk + acq];
    #pragma unroll
    for (int i = 0; i < 4; ++i)
      breg[i] = *(const v4f*)&W[(size_t)(kk + bk + 8 * i) * ld + cn0 + bnq];
  };
  auto store_tiles = [&](int buf) {
    #pragma unroll
    for (int i = 0; i < 4; ++i) {
      v4h h4;
      #pragma unroll
      for (int j = 0; j < 4; ++j) h4[j] = (f16)areg[i][j];
      *(v4h*)&As[buf][(ar + 32 * i) * 32 + acq] = h4;
    }
    #pragma unroll
    for (int i = 0; i < 4; ++i)
      #pragma unroll
      for (int j = 0; j < 4; ++j)
        Bt[buf][(bnq + j) * 32 + bk + 8 * i] = (f16)breg[i][j];
  };

  v8f acc[2][4] = {};
  load_tiles(0);
  store_tiles(0);
  int cur = 0;
  for (int kk = 0; kk < H; kk += 32) {
    __syncthreads();
    const bool more = (kk + 32 < H);
    if (more) load_tiles(kk + 32);      // global loads in flight during WMMA

    const f16* Ac = As[cur];
    const f16* Bc = Bt[cur];
    v16h a[2], b[4];
    #pragma unroll
    for (int im = 0; im < 2; ++im) {
      int r = 32 * wmy + 16 * im + ml;
      a[im] = concat8(*(const v8h*)&Ac[r * 32 + hl * 8],
                      *(const v8h*)&Ac[r * 32 + 16 + hl * 8]);
    }
    #pragma unroll
    for (int in = 0; in < 4; ++in) {
      int n = 64 * wnx + 16 * in + ml;
      b[in] = concat8(*(const v8h*)&Bc[n * 32 + hl * 8],
                      *(const v8h*)&Bc[n * 32 + 16 + hl * 8]);
    }
    #pragma unroll
    for (int im = 0; im < 2; ++im)
      #pragma unroll
      for (int in = 0; in < 4; ++in)
        acc[im][in] = WMMA_F32_F16(a[im], b[in], acc[im][in]);

    if (more) store_tiles(cur ^ 1);
    cur ^= 1;
  }

  // Epilogue: f16 head-major: q[b][h][t][d], k/v[b][kvh][t][d]
  #pragma unroll
  for (int im = 0; im < 2; ++im)
    #pragma unroll
    for (int in = 0; in < 4; ++in)
      #pragma unroll
      for (int r = 0; r < 8; ++r) {
        int gr = m0 + 32 * wmy + 16 * im + r + 8 * hl;
        int gc = n0 + 64 * wnx + 16 * in + ml;
        int bi = gr / T, t = gr % T;
        f16 val = (f16)acc[im][in][r];
        if (gc < 2048) {
          int hd = gc >> 7, d = gc & 127;
          qb[(((size_t)bi * NH + hd) * T + t) * HD + d] = val;
        } else if (gc < 2560) {
          int c = gc - 2048, hd = c >> 7, d = c & 127;
          kb[(((size_t)bi * NKV + hd) * T + t) * HD + d] = val;
        } else {
          int c = gc - 2560, hd = c >> 7, d = c & 127;
          vb[(((size_t)bi * NKV + hd) * T + t) * HD + d] = val;
        }
      }
}

// ---------------------------------------------------------------------------
// Kernel 2: RoPE in place on q or k (pairs d, d+64).
// ---------------------------------------------------------------------------
__global__ __launch_bounds__(256) void rope_kernel(
    f16* __restrict__ buf, const float* __restrict__ cosp,
    const float* __restrict__ sinp, int total) {
  int idx = blockIdx.x * 256 + threadIdx.x;
  if (idx >= total) return;
  int d = idx & 63;
  int t = (idx >> 6) & (T - 1);
  int bh = idx >> 17;                 // T*64 = 2^17
  size_t base = ((size_t)bh * T + t) * HD;
  float x1 = (float)buf[base + d], x2 = (float)buf[base + d + 64];
  float c = cosp[t * 64 + d], s = sinp[t * 64 + d];
  buf[base + d]      = (f16)(x1 * c - x2 * s);
  buf[base + d + 64] = (f16)(x2 * c + x1 * s);
}

// ---------------------------------------------------------------------------
// Kernel 3: causal flash attention.  K tile staged via async-to-LDS copy.
// ---------------------------------------------------------------------------
__global__ __launch_bounds__(256) void attn_kernel(
    const f16* __restrict__ qb, const f16* __restrict__ kb,
    const f16* __restrict__ vb, f16* __restrict__ ob) {
  __shared__ __align__(16) f16 Ks[64 * 128];      // K[key][d] == Bt[n=key][k=d]
  __shared__ __align__(16) f16 Vt[128 * 64];      // V^T[d][key]
  __shared__ __align__(16) f16 Ps[8 * 16 * 64];   // per-wave P strip [16][64]
  const int tid = threadIdx.x, lane = tid & 31, w = tid >> 5;
  const int hl = lane >> 4, ml = lane & 15;
  const int bh = blockIdx.y, b = bh >> 4, h = bh & 15, kvh = h >> 2;
  const int q0 = blockIdx.x * 128;
  const size_t qbase = (size_t)bh * T * HD;
  const size_t kvbase = ((size_t)b * NKV + kvh) * T * HD;
  const float scale = 0.08838834764831845f;   // 1/sqrt(128)

  // Q fragments in registers for the whole kernel
  v16h qf[4];
  {
    const f16* qrow = qb + qbase + (size_t)(q0 + 16 * w + ml) * HD;
    #pragma unroll
    for (int ks = 0; ks < 4; ++ks)
      qf[ks] = concat8(*(const v8h*)&qrow[ks * 32 + hl * 8],
                       *(const v8h*)&qrow[ks * 32 + 16 + hl * 8]);
  }

  v8f oacc[8] = {};
  float mrun[8], lrun[8];
  #pragma unroll
  for (int r = 0; r < 8; ++r) { mrun[r] = -__builtin_inff(); lrun[r] = 0.f; }

  for (int k0 = 0; k0 < q0 + 128; k0 += 64) {
    __syncthreads();
    // K tile: async copy straight to LDS (no VGPR staging)
    #pragma unroll
    for (int i = 0; i < 4; ++i) {
      int chunk = tid + 256 * i, key = chunk >> 4, doff = (chunk & 15) * 8;
      async_copy16(&kb[kvbase + (size_t)(k0 + key) * HD + doff],
                   &Ks[key * 128 + doff]);
    }
    // V tile: wide loads, transposed scatter into LDS
    #pragma unroll
    for (int i = 0; i < 4; ++i) {
      int chunk = tid + 256 * i, key = chunk >> 4, doff = (chunk & 15) * 8;
      v8h vv = *(const v8h*)&vb[kvbase + (size_t)(k0 + key) * HD + doff];
      #pragma unroll
      for (int j = 0; j < 8; ++j) Vt[(doff + j) * 64 + key] = vv[j];
    }
    wait_async0();
    __syncthreads();

    // S = Q @ K^T
    v8f sacc[4] = {};
    #pragma unroll
    for (int ks = 0; ks < 4; ++ks)
      #pragma unroll
      for (int jn = 0; jn < 4; ++jn) {
        int n = 16 * jn + ml;
        v16h bfr = concat8(*(const v8h*)&Ks[n * 128 + ks * 32 + hl * 8],
                           *(const v8h*)&Ks[n * 128 + ks * 32 + 16 + hl * 8]);
        sacc[jn] = WMMA_F32_F16(qf[ks], bfr, sacc[jn]);
      }

    // Scale + causal mask + online softmax
    float rm[8], p[4][8], rs[8];
    #pragma unroll
    for (int r = 0; r < 8; ++r) rm[r] = -__builtin_inff();
    #pragma unroll
    for (int jn = 0; jn < 4; ++jn)
      #pragma unroll
      for (int r = 0; r < 8; ++r) {
        int row = q0 + 16 * w + r + 8 * hl;
        int col = k0 + 16 * jn + ml;
        float s = sacc[jn][r] * scale;
        if (col > row) s = -__builtin_inff();
        p[jn][r] = s;
        rm[r] = fmaxf(rm[r], s);
      }
    #pragma unroll
    for (int r = 0; r < 8; ++r) {
      rm[r] = fmaxf(rm[r], __shfl_xor(rm[r], 1));
      rm[r] = fmaxf(rm[r], __shfl_xor(rm[r], 2));
      rm[r] = fmaxf(rm[r], __shfl_xor(rm[r], 4));
      rm[r] = fmaxf(rm[r], __shfl_xor(rm[r], 8));
      float mn = fmaxf(mrun[r], rm[r]);
      float alpha = __expf(mrun[r] - mn);
      mrun[r] = mn;
      lrun[r] *= alpha;
      #pragma unroll
      for (int jn2 = 0; jn2 < 8; ++jn2) oacc[jn2][r] *= alpha;
      rs[r] = 0.f;
    }
    #pragma unroll
    for (int jn = 0; jn < 4; ++jn)
      #pragma unroll
      for (int r = 0; r < 8; ++r) {
        float e = __expf(p[jn][r] - mrun[r]);
        p[jn][r] = e;
        rs[r] += e;
      }
    #pragma unroll
    for (int r = 0; r < 8; ++r) {
      rs[r] += __shfl_xor(rs[r], 1);
      rs[r] += __shfl_xor(rs[r], 2);
      rs[r] += __shfl_xor(rs[r], 4);
      rs[r] += __shfl_xor(rs[r], 8);
      lrun[r] += rs[r];
    }

    // C-layout -> A-layout for P via per-wave LDS strip
    f16* pw = &Ps[w * 16 * 64];
    #pragma unroll
    for (int jn = 0; jn < 4; ++jn)
      #pragma unroll
      for (int r = 0; r < 8; ++r)
        pw[(r + 8 * hl) * 64 + 16 * jn + ml] = (f16)p[jn][r];
    __syncthreads();

    // O += P @ V
    #pragma unroll
    for (int ks2 = 0; ks2 < 2; ++ks2) {
      v16h af = concat8(*(const v8h*)&pw[ml * 64 + ks2 * 32 + hl * 8],
                        *(const v8h*)&pw[ml * 64 + ks2 * 32 + 16 + hl * 8]);
      #pragma unroll
      for (int jn2 = 0; jn2 < 8; ++jn2) {
        int n = 16 * jn2 + ml;
        v16h bfr = concat8(*(const v8h*)&Vt[n * 64 + ks2 * 32 + hl * 8],
                           *(const v8h*)&Vt[n * 64 + ks2 * 32 + 16 + hl * 8]);
        oacc[jn2] = WMMA_F32_F16(af, bfr, oacc[jn2]);
      }
    }
  }

  // Normalize and write attn output [b][t][h*HD + d] (f16)
  #pragma unroll
  for (int jn2 = 0; jn2 < 8; ++jn2)
    #pragma unroll
    for (int r = 0; r < 8; ++r) {
      int row = q0 + 16 * w + r + 8 * hl;
      int d = 16 * jn2 + ml;
      float val = oacc[jn2][r] / lrun[r];
      ob[((size_t)b * T + row) * H + h * HD + d] = (f16)val;
    }
}

// ---------------------------------------------------------------------------
// Kernel 4: output projection.  A tile (f16) staged via async-to-LDS;
// Wo tile converted f32->f16 through registers.  Double-buffered LDS.
// ---------------------------------------------------------------------------
__global__ __launch_bounds__(256) void out_gemm(
    const f16* __restrict__ A, const float* __restrict__ Wo,
    float* __restrict__ out) {
  __shared__ __align__(16) f16 As[2][128 * 32];
  __shared__ __align__(16) f16 Bt[2][128 * 32];
  const int tid = threadIdx.x, lane = tid & 31, w = tid >> 5;
  const int wmy = w & 3, wnx = w >> 2;
  const int hl = lane >> 4, ml = lane & 15;
  const int m0 = blockIdx.y * 128, n0 = blockIdx.x * 128;
  const int bk = tid >> 5, bnq = (tid & 31) * 4;

  v4f breg[4];
  auto async_loadA = [&](int kk, int buf) {
    #pragma unroll
    for (int i = 0; i < 2; ++i) {
      int chunk = tid + 256 * i, r = chunk >> 2, coff = (chunk & 3) * 8;
      async_copy16(&A[(size_t)(m0 + r) * H + kk + coff],
                   &As[buf][r * 32 + coff]);
    }
  };
  auto loadB = [&](int kk) {
    #pragma unroll
    for (int i = 0; i < 4; ++i)
      breg[i] = *(const v4f*)&Wo[(size_t)(kk + bk + 8 * i) * H + n0 + bnq];
  };
  auto storeB = [&](int buf) {
    #pragma unroll
    for (int i = 0; i < 4; ++i)
      #pragma unroll
      for (int j = 0; j < 4; ++j)
        Bt[buf][(bnq + j) * 32 + bk + 8 * i] = (f16)breg[i][j];
  };

  v8f acc[2][4] = {};
  async_loadA(0, 0);
  loadB(0);
  storeB(0);
  int cur = 0;
  for (int kk = 0; kk < H; kk += 32) {
    wait_async0();
    __syncthreads();
    const bool more = (kk + 32 < H);
    if (more) { async_loadA(kk + 32, cur ^ 1); loadB(kk + 32); }

    const f16* Ac = As[cur];
    const f16* Bc = Bt[cur];
    v16h a[2], b[4];
    #pragma unroll
    for (int im = 0; im < 2; ++im) {
      int r = 32 * wmy + 16 * im + ml;
      a[im] = concat8(*(const v8h*)&Ac[r * 32 + hl * 8],
                      *(const v8h*)&Ac[r * 32 + 16 + hl * 8]);
    }
    #pragma unroll
    for (int in = 0; in < 4; ++in) {
      int n = 64 * wnx + 16 * in + ml;
      b[in] = concat8(*(const v8h*)&Bc[n * 32 + hl * 8],
                      *(const v8h*)&Bc[n * 32 + 16 + hl * 8]);
    }
    #pragma unroll
    for (int im = 0; im < 2; ++im)
      #pragma unroll
      for (int in = 0; in < 4; ++in)
        acc[im][in] = WMMA_F32_F16(a[im], b[in], acc[im][in]);

    if (more) storeB(cur ^ 1);
    cur ^= 1;
  }

  #pragma unroll
  for (int im = 0; im < 2; ++im)
    #pragma unroll
    for (int in = 0; in < 4; ++in)
      #pragma unroll
      for (int r = 0; r < 8; ++r) {
        int gr = m0 + 32 * wmy + 16 * im + r + 8 * hl;
        int gc = n0 + 64 * wnx + 16 * in + ml;
        out[(size_t)gr * H + gc] = acc[im][in][r];
      }
}

// ---------------------------------------------------------------------------
extern "C" void kernel_launch(void* const* d_in, const int* in_sizes, int n_in,
                              void* d_out, int out_size, void* d_ws, size_t ws_size,
                              hipStream_t stream) {
  (void)in_sizes; (void)n_in; (void)out_size; (void)ws_size;
  const float* x    = (const float*)d_in[0];
  const float* cosp = (const float*)d_in[1];
  const float* sinp = (const float*)d_in[2];
  const float* Wq   = (const float*)d_in[3];
  const float* Wk   = (const float*)d_in[4];
  const float* Wv   = (const float*)d_in[5];
  const float* Wo   = (const float*)d_in[6];
  float* out = (float*)d_out;

  char* ws = (char*)d_ws;
  const size_t qbytes = (size_t)Bz * NH * T * HD * sizeof(f16);    // 16 MB
  const size_t kvbytes = (size_t)Bz * NKV * T * HD * sizeof(f16);  // 4 MB
  f16* qb = (f16*)ws;
  f16* kb = (f16*)(ws + qbytes);
  f16* vb = (f16*)(ws + qbytes + kvbytes);
  f16* ab = (f16*)(ws + qbytes + 2 * kvbytes);                     // 16 MB

  qkv_gemm<<<dim3(24, 32), 256, 0, stream>>>(x, Wq, Wk, Wv, qb, kb, vb);
  rope_kernel<<<(Bz * NH * T * 64) / 256, 256, 0, stream>>>(qb, cosp, sinp, Bz * NH * T * 64);
  rope_kernel<<<(Bz * NKV * T * 64) / 256, 256, 0, stream>>>(kb, cosp, sinp, Bz * NKV * T * 64);
  attn_kernel<<<dim3(T / 128, Bz * NH), 256, 0, stream>>>(qb, kb, vb, ab);
  out_gemm<<<dim3(H / 128, 32), 256, 0, stream>>>(ab, Wo, out);
}
